// FFSP_Encoder_75557064671556
// MI455X (gfx1250) — matrix operand order repaired and checked by reading, not compiled, not run on hardware
//
#include <hip/hip_runtime.h>
#include <cstddef>

typedef __attribute__((ext_vector_type(16))) _Float16 v16h;
typedef __attribute__((ext_vector_type(8)))  _Float16 v8h;
typedef __attribute__((ext_vector_type(8)))  float    v8f;
typedef __attribute__((ext_vector_type(2)))  float    v2f;
typedef __attribute__((ext_vector_type(4)))  float    v4f;

// ---------------- problem constants ----------------
constexpr int NB   = 8;     // batch
constexpr int NT   = 256;   // tokens (R == C == 256)
constexpr int DIM  = 256;   // embedding
constexpr int NH   = 16;    // heads
constexpr int HD   = 16;    // head dim
constexpr int FF   = 512;   // ff hidden
constexpr int BN   = NB * NT;               // 2048
constexpr int EMB  = NB * NT * DIM;         // 524288
constexpr int SCR  = NB * NH * NT * NT;     // 8388608
constexpr int HIDS = BN * FF;               // 1048576
constexpr float EPS = 1e-5f;

// float workspace offsets (in floats)
constexpr size_t F_Q    = 0;
constexpr size_t F_K    = EMB;
constexpr size_t F_V    = 2ull * EMB;
constexpr size_t F_ATTN = 3ull * EMB;
constexpr size_t F_X    = 4ull * EMB;
constexpr size_t F_O1   = 5ull * EMB;
constexpr size_t F_HID  = 6ull * EMB;
constexpr size_t F_SC   = F_HID + HIDS;
constexpr size_t F_RA   = F_SC + SCR;
constexpr size_t F_CA   = F_RA + EMB;
constexpr size_t F_RB   = F_CA + EMB;
constexpr size_t F_CB   = F_RB + EMB;
constexpr size_t FLOATS_TOTAL = F_CB + EMB;                 // 14,680,064

// half workspace offsets (in halves), placed after the float region
constexpr size_t H_SCH = 0;                                  // softmax weights f16
constexpr size_t H_VTH = (size_t)SCR;                        // V transposed f16
constexpr size_t H_AH  = H_VTH + EMB;                        // activation staging f16
constexpr size_t H_WH  = H_AH + HIDS;                        // weight staging f16 (transposed)
constexpr size_t HALVES_TOTAL = H_WH + 131072;               // 10,092,544
constexpr size_t WS_BYTES = FLOATS_TOTAL * 4 + HALVES_TOTAL * 2;  // ~75.3 MB

// ---------------- device: 16x16 f32 tile, all-f16 operands -------------------
// A: row-major M x K f16 (lda = K-ish stride). Bt: row-major N x K f16 (W^T).
__device__ __forceinline__ v8f gemm_tile_h(const _Float16* __restrict__ A, int lda,
                                           const _Float16* __restrict__ Bt, int ldb,
                                           int K, int lane)
{
    const int half = lane >> 4;       // 0: lanes 0-15, 1: lanes 16-31
    const int lm   = lane & 15;
    const _Float16* ar = A  + lm * lda;
    const _Float16* br = Bt + lm * ldb;
    v8f acc = {};
    for (int kb = 0; kb < K; kb += 32) {
        // A 16x32: lane<16 -> K 0..7 & 16..23 ; lane>=16 -> K 8..15 & 24..31
        const v8h alo = *(const v8h*)(ar + kb + half * 8);
        const v8h ahi = *(const v8h*)(ar + kb + half * 8 + 16);
        const v16h av = __builtin_shufflevector(alo, ahi,
                0, 1, 2, 3, 4, 5, 6, 7, 8, 9, 10, 11, 12, 13, 14, 15);
        // B 32x16: lane<16 -> K 0..7 & 16..23 ; lane>=16 -> K 8..15 & 24..31 (same as A)
        const v8h blo = *(const v8h*)(br + kb + half * 8);
        const v8h bhi = *(const v8h*)(br + kb + half * 8 + 16);
        const v16h bv = __builtin_shufflevector(blo, bhi,
                0, 1, 2, 3, 4, 5, 6, 7, 8, 9, 10, 11, 12, 13, 14, 15);
        acc = __builtin_amdgcn_wmma_f32_16x16x32_f16(false, av, false, bv,
                                                     (short)0, acc, false, false);
    }
    return acc;
}

// ---------------- GEMM: C[M,N] = A[M,K] @ W[K,N] (+bias)(relu)(+residual) -----
// Ah: M x K f16, Wt: N x K f16 (transposed weights)
__global__ __launch_bounds__(256) void gemm_h_kernel(
    const _Float16* __restrict__ Ah, const _Float16* __restrict__ Wt,
    const float* __restrict__ bias, const float* __restrict__ residual,
    float* __restrict__ C, int M, int N, int K, int relu)
{
    const int tilesN   = N >> 4;
    const int numTiles = (M >> 4) * tilesN;
    const int wave = blockIdx.x * (blockDim.x >> 5) + (threadIdx.x >> 5);
    if (wave >= numTiles) return;              // wave-uniform exit (EXEC stays all-1)
    const int tm = wave / tilesN, tn = wave % tilesN;
    const int lane = threadIdx.x & 31, half = lane >> 4, lm = lane & 15;

    v8f acc = gemm_tile_h(Ah + (size_t)tm * 16 * K, K, Wt + (size_t)tn * 16 * K, K, K, lane);

    const int n = tn * 16 + lm;
    const float badd = bias ? bias[n] : 0.0f;
#pragma unroll
    for (int v = 0; v < 8; ++v) {
        const int m = tm * 16 + v + half * 8;
        float val = acc[v] + badd;
        if (relu) val = fmaxf(val, 0.0f);
        if (residual) val += residual[(size_t)m * N + n];
        C[(size_t)m * N + n] = val;
    }
}

// ---------------- q.kT logits (exact fp32 WMMA, K = head_dim = 16) ------------
__global__ __launch_bounds__(256) void qk_dot_kernel(
    const float* __restrict__ Q, const float* __restrict__ Km, float* __restrict__ S)
{
    const int bh = blockIdx.y;                 // 0..127
    const int b = bh >> 4, h = bh & 15;
    const int wave = blockIdx.x * 8 + (threadIdx.x >> 5);   // 256 tiles per (b,h)
    if (wave >= 256) return;
    const int tm = wave >> 4, tn = wave & 15;
    const int lane = threadIdx.x & 31, half = lane >> 4, lm = lane & 15;

    const float* Aq = Q  + b * (NT * DIM) + (tm * 16) * DIM + h * HD; // [r, d]
    const float* Bk = Km + b * (NT * DIM) + (tn * 16) * DIM + h * HD; // B[k][n]=Bk[n*DIM+k]
    v8f acc = {};
#pragma unroll
    for (int kb = 0; kb < HD; kb += 4) {
        v2f a, bv;
        // f32 A 16x4: VGPR0 = K0 / K2, VGPR1 = K1 / K3 across lane halves
        a[0]  = Aq[lm * DIM + kb + 2 * half];
        a[1]  = Aq[lm * DIM + kb + 2 * half + 1];
        bv[0] = Bk[lm * DIM + kb + 2 * half];
        bv[1] = Bk[lm * DIM + kb + 2 * half + 1];
        acc = __builtin_amdgcn_wmma_f32_16x16x4_f32(false, a, false, bv,
                                                    (short)0, acc, false, false);
    }
    float* Ct = S + (size_t)bh * (NT * NT) + (tm * 16) * NT + tn * 16;
#pragma unroll
    for (int v = 0; v < 8; ++v)
        Ct[(v + half * 8) * NT + lm] = acc[v] * 0.25f;   // 1/sqrt(16)
}

// ---------------- per-score MLP (2->16->1, relu) + softmax over c -------------
// reads fp32 dots from S, writes fp16 softmax weights to Sh
__global__ __launch_bounds__(256) void mlp_softmax_kernel(
    const float* __restrict__ S, _Float16* __restrict__ Sh,
    const float* __restrict__ cost,
    const float* __restrict__ m1w, const float* __restrict__ m1b,
    const float* __restrict__ m2w, const float* __restrict__ m2b, int costT)
{
    __shared__ float red[256];
    const int tid = threadIdx.x;               // c
    const int r = blockIdx.x & 255;
    const int h = (blockIdx.x >> 8) & 15;
    const int b = blockIdx.x >> 12;

    const size_t sidx = (((size_t)(b * NH + h) * NT) + r) * NT + tid;
    const float dot = S[sidx];
    const float cv  = costT ? cost[b * NT * NT + tid * NT + r]
                            : cost[b * NT * NT + r * NT + tid];
    float s = m2b[h];
#pragma unroll
    for (int m = 0; m < 16; ++m) {
        float t = dot * m1w[h * 32 + m] + cv * m1w[h * 32 + 16 + m] + m1b[h * 16 + m];
        t = fmaxf(t, 0.0f);
        s += t * m2w[h * 16 + m];
    }
    red[tid] = s;
    __syncthreads();
    for (int off = 128; off > 0; off >>= 1) {
        if (tid < off) red[tid] = fmaxf(red[tid], red[tid + off]);
        __syncthreads();
    }
    const float mx = red[0];
    __syncthreads();
    const float e = expf(s - mx);
    red[tid] = e;
    __syncthreads();
    for (int off = 128; off > 0; off >>= 1) {
        if (tid < off) red[tid] += red[tid + off];
        __syncthreads();
    }
    Sh[sidx] = (_Float16)(e / red[0]);
}

// ---------------- attention output: out[b,r,h*16+d] = sum_c w * v -------------
// Wh: fp16 softmax weights [bh, r, c]; Vth: fp16 transposed V [bh, d, c]
__global__ __launch_bounds__(256) void attn_out_kernel(
    const _Float16* __restrict__ Wh, const _Float16* __restrict__ Vth,
    float* __restrict__ O)
{
    const int bh = blockIdx.y;
    const int b = bh >> 4, h = bh & 15;
    const int wave = blockIdx.x * 8 + (threadIdx.x >> 5);  // 16 row-tiles, one N tile
    if (wave >= 16) return;
    const int lane = threadIdx.x & 31, half = lane >> 4, lm = lane & 15;

    const _Float16* A  = Wh  + (size_t)bh * (NT * NT) + wave * 16 * NT;
    const _Float16* Bt = Vth + (size_t)bh * (HD * NT);
    v8f acc = gemm_tile_h(A, NT, Bt, NT, NT, lane);

    float* C = O + (size_t)b * (NT * DIM) + wave * 16 * DIM + h * HD;
#pragma unroll
    for (int v = 0; v < 8; ++v)
        C[(v + half * 8) * DIM + lm] = acc[v];
}

// ---------------- InstanceNorm1d over token axis (per b, per channel) ---------
__global__ __launch_bounds__(256) void inorm_kernel(
    const float* __restrict__ X, const float* __restrict__ g,
    const float* __restrict__ be, float* __restrict__ O)
{
    const int b = blockIdx.x, d = threadIdx.x;
    const float* xb = X + b * (NT * DIM) + d;
    float sum = 0.0f, sq = 0.0f;
    for (int n = 0; n < NT; ++n) {
        const float v = xb[n * DIM];
        sum += v; sq += v * v;
    }
    const float mu  = sum * (1.0f / NT);
    const float var = sq * (1.0f / NT) - mu * mu;
    const float sc  = rsqrtf(var + EPS) * g[d];
    const float sh  = be[d] - mu * sc;
    float* ob = O + b * (NT * DIM) + d;
    for (int n = 0; n < NT; ++n)
        ob[n * DIM] = xb[n * DIM] * sc + sh;
}

// ---------------- conversion helpers ------------------------------------------
// fp32 -> fp16, 4 elements per thread (vectorized b128 read)
__global__ __launch_bounds__(256) void cvt_f16_kernel(
    const float* __restrict__ src, _Float16* __restrict__ dst, int n4)
{
    const int i = blockIdx.x * blockDim.x + threadIdx.x;
    if (i >= n4) return;
    const v4f v = *(const v4f*)(src + (size_t)i * 4);
    _Float16* d = dst + (size_t)i * 4;
    d[0] = (_Float16)v.x; d[1] = (_Float16)v.y;
    d[2] = (_Float16)v.z; d[3] = (_Float16)v.w;
}

// W[K,N] fp32 -> Wt[N,K] fp16 (coalesced writes)
__global__ __launch_bounds__(256) void cvt_wt_kernel(
    const float* __restrict__ W, _Float16* __restrict__ Wt, int K, int N)
{
    const int idx = blockIdx.x * blockDim.x + threadIdx.x;
    if (idx >= K * N) return;
    const int n = idx / K, k = idx % K;
    Wt[idx] = (_Float16)W[(size_t)k * N + n];
}

// V[b, c, h*16+d] fp32 -> Vth[(b*NH+h)*HD + d][c] fp16
__global__ __launch_bounds__(256) void cvt_vt_kernel(
    const float* __restrict__ V, _Float16* __restrict__ Vth)
{
    const int idx = blockIdx.x * blockDim.x + threadIdx.x;   // over NB*NH*HD*NT
    const int c = idx & (NT - 1);
    const int d = (idx >> 8) & (HD - 1);
    const int h = (idx >> 12) & (NH - 1);
    const int b = idx >> 16;
    Vth[idx] = (_Float16)V[(size_t)b * NT * DIM + c * DIM + h * HD + d];
}

// ======================= host side ===========================================
// canonical slots: 0 Wq,1 Wk,2 Wv,3 m1w,4 m1b,5 m2w,6 m2b,7 Wo,8 bo,
//                  9 g1,10 be1,11 W1,12 bf1,13 W2,14 bf2,15 g2,16 be2
struct BP { const float* p[17]; };

static const int ALPHA_MAP[17] = { // position in alphabetical order -> canonical slot
    11, 13, 1, 7, 0, 2,            // W1 W2 Wk Wo Wq Wv
    10, 16, 12, 14, 8,             // be1 be2 bf1 bf2 bo
    9, 15,                         // g1 g2
    4, 3, 6, 5                     // m1b m1w m2b m2w
};

static void run_block(const BP& P, const float* x_in, const float* ctx,
                      const float* cost, int costT, float* out,
                      float* wf, _Float16* wh, hipStream_t s)
{
    float* Q    = wf + F_Q;
    float* Kb   = wf + F_K;
    float* V    = wf + F_V;
    float* Attn = wf + F_ATTN;
    float* X    = wf + F_X;
    float* O1   = wf + F_O1;
    float* Hid  = wf + F_HID;
    float* Sc   = wf + F_SC;
    _Float16* Sch = wh + H_SCH;
    _Float16* Vth = wh + H_VTH;
    _Float16* Ah  = wh + H_AH;
    _Float16* Wh  = wh + H_WH;
    const dim3 blk(256);

    // Q = x_in @ Wq ; K/V from ctx (Ah reuse relies on in-stream ordering)
    cvt_f16_kernel<<<EMB / 1024, blk, 0, s>>>(x_in, Ah, EMB / 4);
    cvt_wt_kernel<<<(DIM * DIM) / 256, blk, 0, s>>>(P.p[0], Wh, DIM, DIM);
    gemm_h_kernel<<<256, blk, 0, s>>>(Ah, Wh, nullptr, nullptr, Q, BN, DIM, DIM, 0);
    cvt_f16_kernel<<<EMB / 1024, blk, 0, s>>>(ctx, Ah, EMB / 4);
    cvt_wt_kernel<<<(DIM * DIM) / 256, blk, 0, s>>>(P.p[1], Wh, DIM, DIM);
    gemm_h_kernel<<<256, blk, 0, s>>>(Ah, Wh, nullptr, nullptr, Kb, BN, DIM, DIM, 0);
    cvt_wt_kernel<<<(DIM * DIM) / 256, blk, 0, s>>>(P.p[2], Wh, DIM, DIM);
    gemm_h_kernel<<<256, blk, 0, s>>>(Ah, Wh, nullptr, nullptr, V, BN, DIM, DIM, 0);

    // attention
    qk_dot_kernel<<<dim3(32, NB * NH), blk, 0, s>>>(Q, Kb, Sc);
    mlp_softmax_kernel<<<NB * NH * NT, blk, 0, s>>>(Sc, Sch, cost,
                                                    P.p[3], P.p[4], P.p[5], P.p[6], costT);
    cvt_vt_kernel<<<EMB / 256, blk, 0, s>>>(V, Vth);
    attn_out_kernel<<<dim3(2, NB * NH), blk, 0, s>>>(Sch, Vth, Attn);

    // output projection + residual + norm
    cvt_f16_kernel<<<EMB / 1024, blk, 0, s>>>(Attn, Ah, EMB / 4);
    cvt_wt_kernel<<<(DIM * DIM) / 256, blk, 0, s>>>(P.p[7], Wh, DIM, DIM);
    gemm_h_kernel<<<256, blk, 0, s>>>(Ah, Wh, P.p[8], x_in, X, BN, DIM, DIM, 0);
    inorm_kernel<<<NB, blk, 0, s>>>(X, P.p[9], P.p[10], O1);

    // FFN + residual + norm
    cvt_f16_kernel<<<EMB / 1024, blk, 0, s>>>(O1, Ah, EMB / 4);
    cvt_wt_kernel<<<(DIM * FF) / 256, blk, 0, s>>>(P.p[11], Wh, DIM, FF);
    gemm_h_kernel<<<512, blk, 0, s>>>(Ah, Wh, P.p[12], nullptr, Hid, BN, FF, DIM, 1);
    cvt_f16_kernel<<<HIDS / 1024, blk, 0, s>>>(Hid, Ah, HIDS / 4);
    cvt_wt_kernel<<<(FF * DIM) / 256, blk, 0, s>>>(P.p[13], Wh, FF, DIM);
    gemm_h_kernel<<<256, blk, 0, s>>>(Ah, Wh, P.p[14], O1, X, BN, DIM, FF, 0);
    inorm_kernel<<<NB, blk, 0, s>>>(X, P.p[15], P.p[16], out);
}

extern "C" void kernel_launch(void* const* d_in, const int* in_sizes, int n_in,
                              void* d_out, int out_size, void* d_ws, size_t ws_size,
                              hipStream_t stream)
{
    if (ws_size < WS_BYTES) return;
    if (n_in < 105) return;
    float*    wf = (float*)d_ws;
    _Float16* wh = (_Float16*)(wf + FLOATS_TOTAL);

    // --- decode input ordering ---
    const float *row_in, *col_in, *cost;
    int base, perBlockAlpha, colFirst;
    if (in_sizes[2] == NB * NT * NT) {
        // top-level dict insertion order: row_emb, col_emb, cost_mat, params...
        row_in = (const float*)d_in[0];
        col_in = (const float*)d_in[1];
        cost   = (const float*)d_in[2];
        base = 3;
        if (in_sizes[3] == DIM * DIM) { perBlockAlpha = 0; colFirst = 0; } // insertion (Wq first)
        else                          { perBlockAlpha = 1; colFirst = 1; } // alpha (W1 first)
    } else {
        // full pytree alphabetical: col_emb, cost_mat, params..., row_emb
        col_in = (const float*)d_in[0];
        cost   = (const float*)d_in[1];
        base = 2;
        row_in = (const float*)d_in[104];
        perBlockAlpha = 1; colFirst = 1;
    }

    BP blocks[6];   // [layer*2 + 0] = row block, [layer*2 + 1] = col block
    int idx = base;
    for (int l = 0; l < 3; ++l) {
        const int order[2] = { colFirst ? l * 2 + 1 : l * 2 + 0,
                               colFirst ? l * 2 + 0 : l * 2 + 1 };
        for (int o = 0; o < 2; ++o) {
            BP& bp = blocks[order[o]];
            for (int j = 0; j < 17; ++j) {
                const int canon = perBlockAlpha ? ALPHA_MAP[j] : j;
                bp.p[canon] = (const float*)d_in[idx++];
            }
        }
    }

    float* outRow = (float*)d_out;
    float* outCol = (float*)d_out + EMB;
    const float* cr = row_in;
    const float* cc = col_in;
    for (int l = 0; l < 3; ++l) {
        float *nr, *nc;
        if (l == 2)      { nr = outRow;       nc = outCol; }
        else if (l == 0) { nr = wf + F_RA;    nc = wf + F_CA; }
        else             { nr = wf + F_RB;    nc = wf + F_CB; }
        run_block(blocks[l * 2 + 0], cr, cc, cost, 0, nr, wf, wh, stream);
        run_block(blocks[l * 2 + 1], cc, cr, cost, 1, nc, wf, wh, stream);
        cr = nr; cc = nc;
    }
}
